// MAB_61040075210801
// MI455X (gfx1250) — compile-verified, matrix operand order checked by reading, not run on hardware
//
#include <hip/hip_runtime.h>

// ---------------------------------------------------------------------------
// CDNA5 (gfx1250) WMMA implementation of the MAB block.
// wave32; v_wmma_f32_16x16x32_f16 for all matmuls; f32 accumulation.
// Attention K/V tiles are double-buffered in LDS and moved by the Tensor
// Data Mover (tensor_load_to_lds) with DMA/compute overlap via a
// s_wait_tensorcnt split-barrier pipeline (manual staging fallback kept).
// ---------------------------------------------------------------------------

typedef __attribute__((ext_vector_type(16))) _Float16 v16h;
typedef __attribute__((ext_vector_type(8)))  _Float16 v8h;
typedef __attribute__((ext_vector_type(4)))  _Float16 v4h;
typedef __attribute__((ext_vector_type(8)))  float    v8f;
typedef __attribute__((ext_vector_type(4)))  unsigned int v4u;
typedef __attribute__((ext_vector_type(8)))  int      v8i;
typedef __attribute__((ext_vector_type(4)))  int      v4i;

#define V16CAT(lo, hi) __builtin_shufflevector((lo), (hi), \
    0,1,2,3,4,5,6,7,8,9,10,11,12,13,14,15)

#if __has_builtin(__builtin_amdgcn_tensor_load_to_lds)
#define HAVE_TDM 1
#else
#define HAVE_TDM 0
#endif

static constexpr int BATCH = 16;
static constexpr int NQ    = 1024;
static constexpr int NK    = 1024;
static constexpr int DV    = 512;
static constexpr int NHEAD = 8;
static constexpr int DH    = 64;           // DV / NHEAD
static constexpr int MROWS = BATCH * NQ;   // 16384 rows for the dense GEMMs

// ---------------------------------------------------------------------------
// Dense GEMM:  C[M,N] = epilogue( A[M,K] @ W[K,N] + bias )
// MODE 0: Cf = acc + bias                  (f32 out)
// MODE 1: Cf = resid + relu(acc + bias)    (f32 out)
// MODE 2: Ch = (f16)(acc + bias)           (f16 out, feeds attention/TDM)
// Block: 256 threads (8 waves), tile 128(M) x 64(N), K-step 32.
// Each wave computes a 32x32 sub-tile = 2x2 WMMA 16x16 tiles.
// ---------------------------------------------------------------------------
template<int MODE>
__launch_bounds__(256)
__global__ void gemm_f16_kernel(const float* __restrict__ A,
                                const float* __restrict__ W,
                                const float* __restrict__ bias,
                                float* __restrict__ Cf,
                                _Float16* __restrict__ Ch,
                                const float* __restrict__ resid,
                                int M, int N, int K)
{
    __shared__ _Float16 As[128][40];   // [m][k], padded stride 40 halves (80B)
    __shared__ _Float16 Bs[64][40];    // [n][k] = W transposed, padded

    const int tid  = threadIdx.x;
    const int lane = tid & 31;
    const int wid  = tid >> 5;
    const int wm   = wid & 3;          // 4 wave groups along M (32 rows each)
    const int wn   = wid >> 2;         // 2 wave groups along N (32 cols each)
    const int gm   = blockIdx.y * 128;
    const int gn   = blockIdx.x * 64;

    v8f acc[2][2];
    const v8f vzero = {0.f,0.f,0.f,0.f,0.f,0.f,0.f,0.f};
    #pragma unroll
    for (int i = 0; i < 2; ++i)
        #pragma unroll
        for (int j = 0; j < 2; ++j) acc[i][j] = vzero;

    // global->LDS mapping
    const int arow  = tid >> 1;            // 0..127
    const int ahalf = (tid & 1) * 16;      // 0 or 16 within 32-wide k slab
    const int bn    = tid & 63;            // 0..63  (output col in tile)
    const int bk    = (tid >> 6) * 8;      // 0,8,16,24

    const int l15 = lane & 15;
    const int kb  = (lane >> 4) * 8;       // CDNA5 16-bit A/B fragment K base

    for (int k0 = 0; k0 < K; k0 += 32) {
        // ---- stage A tile (fp32 -> f16) ----
        const float4* ap = (const float4*)(A + (size_t)(gm + arow) * K + k0 + ahalf);
        #pragma unroll
        for (int j = 0; j < 4; ++j) {
            float4 f = ap[j];
            v4h h = { (_Float16)f.x, (_Float16)f.y, (_Float16)f.z, (_Float16)f.w };
            *(v4h*)&As[arow][ahalf + 4*j] = h;
        }
        // ---- stage W tile transposed: Bs[n][k] = W[k0+k][gn+n] ----
        #pragma unroll
        for (int j = 0; j < 8; ++j) {
            float w = W[(size_t)(k0 + bk + j) * N + gn + bn];
            Bs[bn][bk + j] = (_Float16)w;
        }
        __syncthreads();

        // ---- fragments (ISA 16-bit 16x32 layout) ----
        v16h af[2], bf[2];
        #pragma unroll
        for (int i = 0; i < 2; ++i) {
            const int r = wm * 32 + i * 16 + l15;
            v8h lo = *(const v8h*)&As[r][kb];
            v8h hi = *(const v8h*)&As[r][kb + 16];
            af[i] = V16CAT(lo, hi);
        }
        #pragma unroll
        for (int j = 0; j < 2; ++j) {
            const int c = wn * 32 + j * 16 + l15;
            v8h lo = *(const v8h*)&Bs[c][kb];
            v8h hi = *(const v8h*)&Bs[c][kb + 16];
            bf[j] = V16CAT(lo, hi);
        }
        #pragma unroll
        for (int i = 0; i < 2; ++i)
            #pragma unroll
            for (int j = 0; j < 2; ++j)
                acc[i][j] = __builtin_amdgcn_wmma_f32_16x16x32_f16(
                    false, af[i], false, bf[j], (short)0, acc[i][j], false, false);
        __syncthreads();
    }

    // ---- epilogue (C layout: VGPR r -> row (lane>=16?8:0)+r, lane -> col) ----
    const int mhi = (lane >> 4) * 8;
    #pragma unroll
    for (int i = 0; i < 2; ++i) {
        #pragma unroll
        for (int j = 0; j < 2; ++j) {
            const int col = gn + wn * 32 + j * 16 + l15;
            const float bv = bias[col];
            #pragma unroll
            for (int r = 0; r < 8; ++r) {
                const int row = gm + wm * 32 + i * 16 + mhi + r;
                size_t idx = (size_t)row * N + col;
                float v = acc[i][j][r] + bv;
                if constexpr (MODE == 1) v = resid[idx] + fmaxf(v, 0.0f);
                if constexpr (MODE == 2) Ch[idx] = (_Float16)v;
                else                     Cf[idx] = v;
            }
        }
    }
}

#if HAVE_TDM
// Issue one TDM 2D tile load: 32 rows x 64 f16, row stride DV elements,
// HW LDS padding 4 DWORDs after every 32 DWORDs -> 144-byte LDS rows.
__device__ __forceinline__ void tdm_load_tile_f16(unsigned lds_addr,
                                                  const _Float16* gaddr)
{
    const unsigned long long ga = (unsigned long long)gaddr;
    v4u g0 = { 1u,                                   // count=1 (valid D#)
               lds_addr,                             // lds_addr
               (unsigned)(ga & 0xffffffffull),
               (unsigned)((ga >> 32) & 0x01ffffffull) | (2u << 30) }; // type=2
    v8i g1 = { (int)0x07110000,   // data_size=2B, pad_en, interval=32dw, amt=4dw
               (int)(64u << 16),  // abar=0 | tensor_dim0.lo = 64
               (int)(32u << 16),  // tensor_dim0.hi=0 | tensor_dim1.lo = 32
               (int)(64u << 16),  // tensor_dim1.hi=0 | tile_dim0 = 64
               (int)32,           // tile_dim1 = 32 | tile_dim2 = 0
               (int)DV,           // tensor_dim0_stride.lo = 512
               0, 0 };
    v4i gz = { 0, 0, 0, 0 };
#if defined(__clang_major__) && (__clang_major__ >= 23)
    v8i gz8 = { 0,0,0,0,0,0,0,0 };
    __builtin_amdgcn_tensor_load_to_lds(g0, g1, gz, gz, gz8, 0);
#else
    __builtin_amdgcn_tensor_load_to_lds(g0, g1, gz, gz, 0);
#endif
}
#endif

// ---------------------------------------------------------------------------
// Flash attention over one (b, h, 64-query tile). 128 threads = 4 waves,
// each wave owns 16 query rows. Keys processed in chunks of 32, with
// double-buffered LDS tiles and TDM DMA overlapped with WMMA compute:
//   issue(chunk i+1); s_wait_tensorcnt 2  (-> chunk i retired); compute(i)
//   S = (q @ k^T) / sqrt(DV);  A = softmax(S);  O = q + A @ v
// ---------------------------------------------------------------------------
__launch_bounds__(128)
__global__ void attn_kernel(const float* __restrict__ q,
                            const _Float16* __restrict__ k,
                            const _Float16* __restrict__ v,
                            float* __restrict__ o)
{
    __shared__ _Float16 Ks[2][32][72];    // [buf][key][dim], 144B rows
#if HAVE_TDM
    __shared__ _Float16 Vr[2][32][72];    // [buf][key][dim], row-major via TDM
#else
    __shared__ _Float16 Vt[2][64][40];    // [buf][dim][key], manual transpose
#endif
    __shared__ _Float16 Ps[4][16][40];    // per-wave P staging (C->A relayout)

    const int tid  = threadIdx.x;
    const int lane = tid & 31;
    const int wave = tid >> 5;
    const int b    = blockIdx.z;
    const int h    = blockIdx.y;
    const int q0   = blockIdx.x * 64 + wave * 16;
    const int l15  = lane & 15;
    const int kb   = (lane >> 4) * 8;
    const float scale = 0.044194173824159216f;   // 1/sqrt(512)

    const size_t headoff = (size_t)h * DH;
    const float*    qb = q + (size_t)b * NQ * DV + headoff;
    const _Float16* kp = k + (size_t)b * NK * DV + headoff;
    const _Float16* vp = v + (size_t)b * NK * DV + headoff;
    float*          op = o + (size_t)b * NQ * DV + headoff;

    // ---- Q fragments (16 rows x 64 dims -> 2 x v16h), fp32->f16 ----
    v16h aq[2];
    {
        const float* qr = qb + (size_t)(q0 + l15) * DV;
        #pragma unroll
        for (int f = 0; f < 2; ++f) {
            float4 a0 = *(const float4*)(qr + f * 32 + kb);
            float4 a1 = *(const float4*)(qr + f * 32 + kb + 4);
            float4 c0 = *(const float4*)(qr + f * 32 + kb + 16);
            float4 c1 = *(const float4*)(qr + f * 32 + kb + 20);
            v8h lo = { (_Float16)a0.x, (_Float16)a0.y, (_Float16)a0.z, (_Float16)a0.w,
                       (_Float16)a1.x, (_Float16)a1.y, (_Float16)a1.z, (_Float16)a1.w };
            v8h hi = { (_Float16)c0.x, (_Float16)c0.y, (_Float16)c0.z, (_Float16)c0.w,
                       (_Float16)c1.x, (_Float16)c1.y, (_Float16)c1.z, (_Float16)c1.w };
            aq[f] = V16CAT(lo, hi);
        }
    }

    v8f oacc[4];
    const v8f vzero = {0.f,0.f,0.f,0.f,0.f,0.f,0.f,0.f};
    #pragma unroll
    for (int d = 0; d < 4; ++d) oacc[d] = vzero;
    float mrun[8], lrun[8];
    #pragma unroll
    for (int r = 0; r < 8; ++r) { mrun[r] = -3.0e38f; lrun[r] = 0.0f; }

#if HAVE_TDM
    const unsigned ldsK = (unsigned)(unsigned long long)(&Ks[0][0][0]);
    const unsigned ldsV = (unsigned)(unsigned long long)(&Vr[0][0][0]);
    const unsigned bufBytes = 32u * 72u * 2u;    // 4608B per buffer
#else
    const int lkey = tid >> 2;            // 0..31
    const int lseg = (tid & 3) * 16;      // 0,16,32,48 (halves)
#endif

    // stage(chunk): move K/V 32x64 f16 tiles for `chunk` into buf[chunk&1]
    auto stage = [&](int chunk) {
#if HAVE_TDM
        if (wave == 0) {
            const unsigned bsel = (unsigned)(chunk & 1) * bufBytes;
            tdm_load_tile_f16(ldsK + bsel, kp + (size_t)chunk * 32 * DV);
            tdm_load_tile_f16(ldsV + bsel, vp + (size_t)chunk * 32 * DV);
        }
#else
        const int bsel = chunk & 1;
        const _Float16* kr = kp + (size_t)(chunk * 32 + lkey) * DV + lseg;
        *(v8h*)&Ks[bsel][lkey][lseg]     = *(const v8h*)(kr);
        *(v8h*)&Ks[bsel][lkey][lseg + 8] = *(const v8h*)(kr + 8);
        const _Float16* vr = vp + (size_t)(chunk * 32 + lkey) * DV + lseg;
        v8h v0 = *(const v8h*)(vr);
        v8h v1 = *(const v8h*)(vr + 8);
        #pragma unroll
        for (int j = 0; j < 8; ++j) {
            Vt[bsel][lseg + j][lkey]     = v0[j];
            Vt[bsel][lseg + 8 + j][lkey] = v1[j];
        }
#endif
    };

    const int NCH = NK / 32;              // 32 key chunks
    stage(0);

    for (int ch = 0; ch < NCH; ++ch) {
        if (ch + 1 < NCH) stage(ch + 1);  // DMA for next chunk, other buffer
#if HAVE_TDM
        if (wave == 0) {
            // in-order TENSORcnt: <=2 outstanding means chunk `ch` retired
#if __has_builtin(__builtin_amdgcn_s_wait_tensorcnt)
            if (ch + 1 < NCH) __builtin_amdgcn_s_wait_tensorcnt(2);
            else              __builtin_amdgcn_s_wait_tensorcnt(0);
#endif
        }
#endif
        __syncthreads();                  // chunk `ch` visible to all waves
        const int pb = ch & 1;

        // ---- scores: two 16-key subtiles, dh=64 -> 2 WMMAs each ----
        v8f s[2];
        #pragma unroll
        for (int st = 0; st < 2; ++st) {
            s[st] = vzero;
            #pragma unroll
            for (int f = 0; f < 2; ++f) {
                v8h lo = *(const v8h*)&Ks[pb][st * 16 + l15][f * 32 + kb];
                v8h hi = *(const v8h*)&Ks[pb][st * 16 + l15][f * 32 + kb + 16];
                v16h bk_ = V16CAT(lo, hi);
                s[st] = __builtin_amdgcn_wmma_f32_16x16x32_f16(
                    false, aq[f], false, bk_, (short)0, s[st], false, false);
            }
        }

        // ---- online softmax over the 32-key chunk ----
        float mnew[8];
        #pragma unroll
        for (int r = 0; r < 8; ++r) {
            float a = s[0][r] * scale;
            float c = s[1][r] * scale;
            s[0][r] = a; s[1][r] = c;
            float mt = fmaxf(a, c);
            mt = fmaxf(mt, __shfl_xor(mt, 1, 32));
            mt = fmaxf(mt, __shfl_xor(mt, 2, 32));
            mt = fmaxf(mt, __shfl_xor(mt, 4, 32));
            mt = fmaxf(mt, __shfl_xor(mt, 8, 32));
            mnew[r] = fmaxf(mrun[r], mt);
        }
        #pragma unroll
        for (int r = 0; r < 8; ++r) {
            const float alpha = __expf(mrun[r] - mnew[r]);
            mrun[r] = mnew[r];
            float p0 = __expf(s[0][r] - mnew[r]);
            float p1 = __expf(s[1][r] - mnew[r]);
            s[0][r] = p0; s[1][r] = p1;
            float rs = p0 + p1;
            rs += __shfl_xor(rs, 1, 32);
            rs += __shfl_xor(rs, 2, 32);
            rs += __shfl_xor(rs, 4, 32);
            rs += __shfl_xor(rs, 8, 32);
            lrun[r] = lrun[r] * alpha + rs;
            #pragma unroll
            for (int d = 0; d < 4; ++d) oacc[d][r] *= alpha;
        }

        // ---- P: C layout -> LDS -> A fragment layout ----
        const int prow = (lane >> 4) * 8;
        #pragma unroll
        for (int r = 0; r < 8; ++r) {
            Ps[wave][prow + r][l15]      = (_Float16)s[0][r];
            Ps[wave][prow + r][16 + l15] = (_Float16)s[1][r];
        }
        asm volatile("s_wait_dscnt 0" ::: "memory");   // same-wave LDS RAW
        v8h plo = *(const v8h*)&Ps[wave][l15][kb];
        v8h phi = *(const v8h*)&Ps[wave][l15][kb + 16];
        v16h ap = V16CAT(plo, phi);

        // ---- O += P @ V  (4 output dim tiles) ----
        #pragma unroll
        for (int d = 0; d < 4; ++d) {
            v16h bv_;
#if HAVE_TDM
            // gather B fragment from row-major Vr: halves j -> key kb+j / 16+kb+j
            #pragma unroll
            for (int j = 0; j < 8; ++j) {
                bv_[j]     = Vr[pb][kb + j][d * 16 + l15];
                bv_[j + 8] = Vr[pb][16 + kb + j][d * 16 + l15];
            }
#else
            v8h blo = *(const v8h*)&Vt[pb][d * 16 + l15][kb];
            v8h bhi = *(const v8h*)&Vt[pb][d * 16 + l15][kb + 16];
            bv_ = V16CAT(blo, bhi);
#endif
            oacc[d] = __builtin_amdgcn_wmma_f32_16x16x32_f16(
                false, ap, false, bv_, (short)0, oacc[d], false, false);
        }
        __syncthreads();   // all waves done with buf[pb] before its reuse
    }

    // ---- normalize, add projected-q residual, store ----
    const int mhi = (lane >> 4) * 8;
    #pragma unroll
    for (int r = 0; r < 8; ++r) {
        const int row = q0 + mhi + r;
        const float inv = 1.0f / lrun[r];
        #pragma unroll
        for (int d = 0; d < 4; ++d) {
            const int col = d * 16 + l15;
            const float qres = qb[(size_t)row * DV + col];
            op[(size_t)row * DV + col] = oacc[d][r] * inv + qres;
        }
    }
}

// ---------------------------------------------------------------------------
// Row-wise LayerNorm over DV=512. One block (256 threads) per row.
// ---------------------------------------------------------------------------
__launch_bounds__(256)
__global__ void ln_kernel(const float* __restrict__ x,
                          const float* __restrict__ g,
                          const float* __restrict__ bt,
                          float* __restrict__ y)
{
    const int row = blockIdx.x;
    const int tid = threadIdx.x;
    const float a0 = x[(size_t)row * DV + tid];
    const float a1 = x[(size_t)row * DV + 256 + tid];
    float s = a0 + a1;
    float qq = a0 * a0 + a1 * a1;
    #pragma unroll
    for (int off = 16; off > 0; off >>= 1) {
        s  += __shfl_xor(s, off, 32);
        qq += __shfl_xor(qq, off, 32);
    }
    __shared__ float ss[8], sq[8];
    if ((tid & 31) == 0) { ss[tid >> 5] = s; sq[tid >> 5] = qq; }
    __syncthreads();
    float ts = 0.f, tq = 0.f;
    #pragma unroll
    for (int i = 0; i < 8; ++i) { ts += ss[i]; tq += sq[i]; }
    const float mu   = ts * (1.0f / 512.0f);
    const float var  = tq * (1.0f / 512.0f) - mu * mu;
    const float rstd = rsqrtf(var + 1e-5f);
    y[(size_t)row * DV + tid]       = (a0 - mu) * rstd * g[tid]       + bt[tid];
    y[(size_t)row * DV + 256 + tid] = (a1 - mu) * rstd * g[256 + tid] + bt[256 + tid];
}

// ---------------------------------------------------------------------------
extern "C" void kernel_launch(void* const* d_in, const int* in_sizes, int n_in,
                              void* d_out, int out_size, void* d_ws, size_t ws_size,
                              hipStream_t stream)
{
    (void)in_sizes; (void)n_in; (void)out_size; (void)ws_size;
    const float* Q  = (const float*)d_in[0];
    const float* K  = (const float*)d_in[1];
    const float* Wq = (const float*)d_in[2];
    const float* bq = (const float*)d_in[3];
    const float* Wk = (const float*)d_in[4];
    const float* bk = (const float*)d_in[5];
    const float* Wv = (const float*)d_in[6];
    const float* bv = (const float*)d_in[7];
    const float* Wo = (const float*)d_in[8];
    const float* bo = (const float*)d_in[9];
    const float* g0 = (const float*)d_in[10];
    const float* b0 = (const float*)d_in[11];
    const float* g1 = (const float*)d_in[12];
    const float* b1 = (const float*)d_in[13];
    float* out = (float*)d_out;

    float* ws = (float*)d_ws;
    const size_t SZ = (size_t)MROWS * DV;        // 8M elements
    float*    qbuf = ws;                         // f32, 32 MB
    float*    obuf = ws + SZ;                    // f32, 32 MB
    float*    t0   = ws + 2 * SZ;                // f32, 32 MB
    float*    t1   = ws + 3 * SZ;                // f32, 32 MB
    _Float16* kh   = (_Float16*)(ws + 4 * SZ);   // f16, 16 MB
    _Float16* vh   = kh + SZ;                    // f16, 16 MB

    const dim3 ggrid(DV / 64, MROWS / 128);      // (8, 128)

    // Projections: q (f32 for residual) ; k, v stored as f16 for attention
    gemm_f16_kernel<0><<<ggrid, 256, 0, stream>>>(Q, Wq, bq, qbuf, nullptr, nullptr, MROWS, DV, DV);
    gemm_f16_kernel<2><<<ggrid, 256, 0, stream>>>(K, Wk, bk, nullptr, kh, nullptr, MROWS, DV, DV);
    gemm_f16_kernel<2><<<ggrid, 256, 0, stream>>>(K, Wv, bv, nullptr, vh, nullptr, MROWS, DV, DV);

    // O = q + softmax(q k^T / sqrt(DV)) v
    attn_kernel<<<dim3(NQ / 64, NHEAD, BATCH), 128, 0, stream>>>(qbuf, kh, vh, obuf);

    // O = LN0(O)
    ln_kernel<<<MROWS, 256, 0, stream>>>(obuf, g0, b0, t0);

    // O = O + relu(O @ Wo + bo)
    gemm_f16_kernel<1><<<ggrid, 256, 0, stream>>>(t0, Wo, bo, t1, nullptr, t0, MROWS, DV, DV);

    // out = LN1(O)
    ln_kernel<<<MROWS, 256, 0, stream>>>(t1, g1, b1, out);
}
